// ModulatedConv2DLayer_47132971106512
// MI455X (gfx1250) — matrix operand order repaired and checked by reading, not compiled, not run on hardware
//
#include <hip/hip_runtime.h>

// ---------------------------------------------------------------------------
// StyleGAN2 modulated conv2d for gfx1250 (MI455X).
// bf16 implicit-GEMM conv: async global->LDS copies (ASYNCcnt), double-buffered
// LDS, v_wmma_f32_16x16x32_bf16 with f32 accumulation.
//   N=8, Cin=Cout=512, H=W=64, K=3x3 SAME.
// All transposes are done once in prep kernels so the hot loop is pure
// 16B-aligned DMA + WMMA.
// ---------------------------------------------------------------------------

typedef __attribute__((ext_vector_type(16))) __bf16 v16bf;
typedef __attribute__((ext_vector_type(8)))  __bf16 bf8;
typedef __attribute__((ext_vector_type(8)))  float  v8f;

#define CIN    512
#define COUT   512
#define LATENT 512
#define NBATCH 8
#define HWDIM  64
#define SPAT   4096
#define TAPS   9
#define KCH    32                 // K-chunk per GEMM step
#define NKC    (CIN / KCH)        // 16 chunks per tap
#define NITER  (TAPS * NKC)       // 144 K-steps

static constexpr float RC_W = 0.0147313912747197390f;  // 1/sqrt(9*512)
static constexpr float RC_S = 0.0441941738241592160f;  // 1/sqrt(512)

__device__ __forceinline__ __bf16 f2bf(float f) {
  unsigned u = __builtin_bit_cast(unsigned, f);
  u += 0x7FFFu + ((u >> 16) & 1u);                 // round to nearest even
  unsigned short h = (unsigned short)(u >> 16);
  return __builtin_bit_cast(__bf16, h);
}

// ---------------- kernel 1: styles s[n][ci] ----------------
__global__ void style_kernel(const float* __restrict__ dl,
                             const float* __restrict__ mw,
                             const float* __restrict__ mb,
                             float* __restrict__ s) {
  int ci = blockIdx.x * blockDim.x + threadIdx.x;
  int n  = blockIdx.y;
  float acc = 0.f;
  const float* dln = dl + n * LATENT;
  for (int l = 0; l < LATENT; ++l)
    acc += dln[l] * mw[l * CIN + ci];
  s[n * CIN + ci] = acc * RC_S + mb[ci] + 1.0f;
}

// ---------------- kernel 2: x (NCHW f32) -> xbT [n][hw][ci] bf16 ----------
// 32x32 LDS tile transpose; both global phases fully coalesced.
__global__ void xcastT_kernel(const float* __restrict__ x,
                              __bf16* __restrict__ xbT) {
  __shared__ __bf16 t[32][33];
  int n  = blockIdx.z;
  int cb = blockIdx.y * 32;
  int hb = blockIdx.x * 32;
  int tx = threadIdx.x, ty = threadIdx.y;   // (32, 8)
  #pragma unroll
  for (int i = 0; i < 4; ++i) {
    int ci = ty + i * 8;
    t[ci][tx] = f2bf(x[((size_t)(n * CIN + cb + ci) << 12) + hb + tx]);
  }
  __syncthreads();
  #pragma unroll
  for (int i = 0; i < 4; ++i) {
    int hw = ty + i * 8;
    xbT[(size_t)(n * SPAT + hb + hw) * CIN + cb + tx] = t[tx][hw];
  }
}

// ---------------- kernel 3: modulate + demod -> wgT ----------------
// wgT layout: [n][tap][cc][co][k32]  (A-fragment friendly; 16B vector stores)
__global__ void modw_kernel(const float* __restrict__ w,   // [9][CIN][COUT]
                            const float* __restrict__ s,   // [N][CIN]
                            __bf16* __restrict__ wgT) {
  __shared__ float sh_s[CIN];
  int n  = blockIdx.y;
  int co = blockIdx.x * blockDim.x + threadIdx.x;
  for (int i = threadIdx.x; i < CIN; i += blockDim.x)
    sh_s[i] = s[n * CIN + i];
  __syncthreads();
  float sum = 0.f;
  for (int r = 0; r < TAPS; ++r)
    for (int ci = 0; ci < CIN; ++ci) {
      float m = w[(r * CIN + ci) * COUT + co] * RC_W * sh_s[ci];
      sum += m * m;
    }
  float d = __frsqrt_rn(sum + 1e-8f);
  for (int r = 0; r < TAPS; ++r)
    for (int cc = 0; cc < NKC; ++cc)
      #pragma unroll
      for (int k8 = 0; k8 < KCH; k8 += 8) {
        bf8 pk;
        #pragma unroll
        for (int q = 0; q < 8; ++q) {
          int ci = cc * KCH + k8 + q;
          float m = w[(r * CIN + ci) * COUT + co] * RC_W * sh_s[ci];
          pk[q] = f2bf(m * d);
        }
        *(bf8*)&wgT[(((size_t)(n * TAPS + r) * NKC + cc) * COUT + co) * KCH + k8] = pk;
      }
}

// ---------------- kernel 4: implicit-GEMM modulated conv ----------------
// D[co][hw] = sum_{tap,ci} Wmod[tap][ci][co] * X[ci][hw+shift]
// Block tile 128(cout) x 128(spatial), K-step 32; 8 waves = 2(M) x 4(N),
// wave tile 64x32 -> 4x2 wmma_f32_16x16x32_bf16 accumulators.
// Tiles staged with global_load_async_to_lds_b128, double-buffered.
#define BM   128
#define BN   128
#define LSTR 40   // LDS row stride in bf16: 80B = 16B aligned, conflict-free

__global__ void __launch_bounds__(256, 2)
modconv_kernel(const __bf16* __restrict__ xbT,   // [N][hw][CIN]
               const __bf16* __restrict__ wgT,   // [N][9][16][COUT][32]
               float* __restrict__ out) {        // [N][COUT][64][64]
  __shared__ __bf16 Wlds[2][BM * LSTR];
  __shared__ __bf16 Xlds[2][BN * LSTR];

  const int tid   = threadIdx.x;
  const int lane  = tid & 31;
  const int wid   = tid >> 5;
  const int waveM = wid & 1;      // 2 waves over cout, 64 rows each
  const int waveN = wid >> 1;     // 4 waves over spatial, 32 cols each
  const int lr      = lane & 15;
  const int halfsel = lane >> 4;

  const int n      = blockIdx.z;
  const int cobase = blockIdx.x * BM;
  const int mbase  = blockIdx.y * BN;

  // per-thread staging coordinates: 2 chunks of 8 bf16 per tile
  const int chunk0 = tid * 2;
  const int row0   = chunk0 >> 2;            // 0..127 (1 row per thread pair)
  const int k8_0   = (chunk0 & 3) << 3;      // 0/16

  v8f acc[4][2];
  #pragma unroll
  for (int i = 0; i < 4; ++i)
    #pragma unroll
    for (int j = 0; j < 2; ++j)
      #pragma unroll
      for (int q = 0; q < 8; ++q) acc[i][j][q] = 0.f;

  auto stage = [&](int it, int buf) {
    const int r  = it >> 4;                  // tap 0..8
    const int cc = it & 15;                  // ci chunk
    const int dh = r / 3 - 1;
    const int dw = r % 3 - 1;
    const __bf16* wsrc =
        wgT + (((size_t)(n * TAPS + r) * NKC + cc) * COUT + cobase) * KCH;
    // ---- W tile: 128 rows(co) x 32k, pure async copy ----
    #pragma unroll
    for (int j = 0; j < 2; ++j) {
      int chunk = chunk0 + j;
      int co = chunk >> 2, k8 = (chunk & 3) << 3;
      unsigned long long ga = (unsigned long long)(wsrc + (size_t)co * KCH + k8);
      unsigned la = (unsigned)(size_t)&Wlds[buf][co * LSTR + k8];
      asm volatile("global_load_async_to_lds_b128 %0, %1, off"
                   :: "v"(la), "v"(ga) : "memory");
    }
    // ---- X tile: 128 rows(m) x 32k; tap shift is row-granular ----
    #pragma unroll
    for (int j = 0; j < 2; ++j) {
      int chunk = chunk0 + j;
      int m = chunk >> 2, k8 = (chunk & 3) << 3;
      int hw = mbase + m;
      int h = (hw >> 6) + dh;
      int w = (hw & 63) + dw;
      unsigned la = (unsigned)(size_t)&Xlds[buf][m * LSTR + k8];
      if ((unsigned)h < 64u && (unsigned)w < 64u) {
        unsigned long long ga = (unsigned long long)(
            xbT + (size_t)(n * SPAT + (h << 6) + w) * CIN + cc * KCH + k8);
        asm volatile("global_load_async_to_lds_b128 %0, %1, off"
                     :: "v"(la), "v"(ga) : "memory");
      } else {
        *(uint4*)&Xlds[buf][m * LSTR + k8] = uint4{0u, 0u, 0u, 0u};
      }
    }
  };

  stage(0, 0);

  for (int it = 0; it < NITER; ++it) {
    const int buf = it & 1;
    asm volatile("s_wait_asynccnt 0x0" ::: "memory");  // this buf's DMA done
    __syncthreads();                                   // visible to all waves
    if (it + 1 < NITER) stage(it + 1, buf ^ 1);        // overlap next DMA

    // A fragments (weights, 16x32): lanes0-15 K{0..7,16..23}, lanes16-31 +8
    v16bf afrag[4], bfrag[2];
    #pragma unroll
    for (int fm = 0; fm < 4; ++fm) {
      const __bf16* p = &Wlds[buf][(waveM * 64 + fm * 16 + lr) * LSTR + halfsel * 8];
      bf8 lo = *(const bf8*)p;
      bf8 hi = *(const bf8*)(p + 16);
      #pragma unroll
      for (int q = 0; q < 8; ++q) { afrag[fm][q] = lo[q]; afrag[fm][q + 8] = hi[q]; }
    }
    // B fragments (x, 32x16): lane = spatial column, lanes0-15 K{0..7,16..23}, lanes16-31 +8
    #pragma unroll
    for (int fn = 0; fn < 2; ++fn) {
      const __bf16* p = &Xlds[buf][(waveN * 32 + fn * 16 + lr) * LSTR + halfsel * 8];
      bf8 lo = *(const bf8*)p;
      bf8 hi = *(const bf8*)(p + 16);
      #pragma unroll
      for (int q = 0; q < 8; ++q) { bfrag[fn][q] = lo[q]; bfrag[fn][q + 8] = hi[q]; }
    }

    #pragma unroll
    for (int fm = 0; fm < 4; ++fm)
      #pragma unroll
      for (int fn = 0; fn < 2; ++fn)
        acc[fm][fn] = __builtin_amdgcn_wmma_f32_16x16x32_bf16(
            false, afrag[fm], false, bfrag[fn],
            (short)0, acc[fm][fn], false, false);
  }

  // writeback: lanes 0-15 span contiguous spatial -> 64B coalesced stores
  #pragma unroll
  for (int fm = 0; fm < 4; ++fm) {
    #pragma unroll
    for (int fn = 0; fn < 2; ++fn) {
      int m = mbase + waveN * 32 + fn * 16 + lr;
      #pragma unroll
      for (int j = 0; j < 8; ++j) {
        int co = cobase + waveM * 64 + fm * 16 + j + halfsel * 8;
        out[((size_t)(n * COUT + co) << 12) + m] = acc[fm][fn][j];
      }
    }
  }
}

// ---------------------------------------------------------------------------
extern "C" void kernel_launch(void* const* d_in, const int* in_sizes, int n_in,
                              void* d_out, int out_size, void* d_ws, size_t ws_size,
                              hipStream_t stream) {
  const float* x  = (const float*)d_in[0];   // [8][512][64][64]
  const float* dl = (const float*)d_in[1];   // [8][512]
  const float* w  = (const float*)d_in[2];   // [3][3][512][512]
  const float* mw = (const float*)d_in[3];   // [512][512]
  const float* mb = (const float*)d_in[4];   // [512]
  float* out = (float*)d_out;

  // workspace: styles (16KB) | xbT bf16 (33.6MB) | wgT bf16 (37.7MB) ~= 71.3MB
  char* ws = (char*)d_ws;
  float*  s   = (float*)ws;
  __bf16* xbT = (__bf16*)(ws + 16384);
  __bf16* wgT = (__bf16*)(ws + 16384 + (size_t)NBATCH * CIN * SPAT * 2);

  style_kernel<<<dim3(CIN / 256, NBATCH), 256, 0, stream>>>(dl, mw, mb, s);
  xcastT_kernel<<<dim3(SPAT / 32, CIN / 32, NBATCH), dim3(32, 8), 0, stream>>>(x, xbT);
  modw_kernel<<<dim3(COUT / 256, NBATCH), 256, 0, stream>>>(w, s, wgT);
  modconv_kernel<<<dim3(COUT / BM, SPAT / BN, NBATCH), 256, 0, stream>>>(xbT, wgT, out);
}
